// Decoder_56495999811733
// MI455X (gfx1250) — matrix operand order repaired and checked by reading, not compiled, not run on hardware
//
#include <hip/hip_runtime.h>
#include <math.h>

// Problem constants (from reference setup_inputs)
#define NN    50000
#define CC    128
#define ES_N  200000
#define EM_N  1000000
#define H0_N  256
#define H1_N  128

typedef __attribute__((ext_vector_type(16))) __bf16          v16bf;
typedef __attribute__((ext_vector_type(16))) unsigned short  v16u;
typedef __attribute__((ext_vector_type(8)))  unsigned short  v8u;
typedef __attribute__((ext_vector_type(8)))  float           v8f;

union BFV { v16u u; v16bf b; };

__device__ __forceinline__ unsigned short f2bf(float f) {
  unsigned int x = __float_as_uint(f);
  x += 0x7fffu + ((x >> 16) & 1u);      // round-to-nearest-even
  return (unsigned short)(x >> 16);
}

__device__ __forceinline__ v8f zero8() {
  v8f z;
#pragma unroll
  for (int i = 0; i < 8; ++i) z[i] = 0.0f;
  return z;
}

__device__ __forceinline__ float wave_sum(float v) {
#pragma unroll
  for (int off = 16; off > 0; off >>= 1) v += __shfl_down(v, off);
  return v;   // full sum in lane 0 (wave32)
}

// ---------------------------------------------------------------------------
// Pack a row-major f32 weight [K x Nd] into bf16 WMMA B-fragment tiles.
// Tile (kt, nt): 32 lanes x 16 bf16 contiguous per lane.
// Lane l, element j -> K = kt*32 + (l<16?0:8) + (j&7) + (j>=8?16:0), N = nt*16 + (l&15).
// ---------------------------------------------------------------------------
__global__ __launch_bounds__(256) void k_pack(const float* __restrict__ W, int Nd,
                                              int ntiles, int total,
                                              unsigned short* __restrict__ outp) {
  int i = blockIdx.x * 256 + threadIdx.x;
  if (i >= total) return;
  int j = i & 15;
  int l = (i >> 4) & 31;
  int tile = i >> 9;
  int nt = tile % ntiles;
  int kt = tile / ntiles;
  int k = kt * 32 + ((l < 16) ? 0 : 8) + (j & 7) + ((j >> 3) << 4);
  int n = nt * 16 + (l & 15);
  outp[i] = f2bf(W[(size_t)k * Nd + n]);
}

// coeff = softplus(raw) / sum(softplus(raw)); also written to output tail
__global__ void k_coeff(const float* __restrict__ raw, float* __restrict__ coeff_ws,
                        float* __restrict__ out) {
  if (threadIdx.x == 0 && blockIdx.x == 0) {
    float u[3], s = 0.f;
#pragma unroll
    for (int i = 0; i < 3; ++i) {
      float x = raw[i];
      u[i] = (x > 20.f) ? x : log1pf(__expf(x));
      s += u[i];
    }
#pragma unroll
    for (int i = 0; i < 3; ++i) {
      float c = u[i] / s;
      coeff_ws[i] = c;
      out[(size_t)4 * ES_N + i] = c;
    }
  }
}

__global__ __launch_bounds__(256) void k_init(int* __restrict__ partner,
                                              float* __restrict__ denom,
                                              float* __restrict__ cntf,
                                              float* __restrict__ numer) {
  int i = blockIdx.x * 256 + threadIdx.x;
  if (i < NN) partner[i] = -1;
  if (i < NN + 1) { denom[i] = 0.f; cntf[i] = 0.f; numer[i] = 0.f; }
}

// inv[n] = 1 / max(||nodes[n]||, 1e-8)  (one wave per node row)
__global__ __launch_bounds__(256) void k_unitinv(const float* __restrict__ nodes,
                                                 float* __restrict__ inv) {
  int t = blockIdx.x * 256 + threadIdx.x;
  int n = t >> 5;
  int lane = t & 31;
  if (n >= NN) return;
  float4 x = *(const float4*)(nodes + (size_t)n * CC + lane * 4);
  float p = x.x * x.x + x.y * x.y + x.z * x.z + x.w * x.w;
  p = wave_sum(p);
  if (lane == 0) inv[n] = 1.0f / fmaxf(sqrtf(p), 1e-8f);
}

// partner_index[sup_partner] = max(arange)  (last-write-wins == max of arange)
__global__ __launch_bounds__(256) void k_scatter(const int* __restrict__ sup_src,
                                                 const int* __restrict__ sup_dst,
                                                 int* __restrict__ partner) {
  int i = blockIdx.x * 256 + threadIdx.x;
  if (i >= 2 * ES_N) return;
  int p = (i < ES_N) ? sup_dst[i] : sup_src[i - ES_N];   // sup_partner = [dst, src]
  atomicMax(&partner[p], i);
}

// One wave per message edge: q, cos, and softmax denominator/count atomics.
// Fixed shift of INTENSITY=10 (score <= 10) replaces the per-segment max:
// exp(s - 10)/sum exp(s - 10) is identical to the reference softmax.
__global__ __launch_bounds__(256) void k_cos(const float* __restrict__ nodes,
                                             const float* __restrict__ inv,
                                             const int* __restrict__ msg_src,
                                             const int* __restrict__ msg_dst,
                                             const int* __restrict__ partner,
                                             const int* __restrict__ sup_src,
                                             const int* __restrict__ sup_dst,
                                             int* __restrict__ qv,
                                             float* __restrict__ cosv,
                                             float* __restrict__ denom,
                                             float* __restrict__ cntf) {
  int t = blockIdx.x * 256 + threadIdx.x;
  int e = t >> 5;                 // wave-uniform edge id
  int lane = t & 31;
  if (e >= EM_N) return;
  int md = msg_dst[e];
  int pidx = partner[md];
  if (pidx < 0) {                 // invalid edges feed only the dumped segment N
    if (lane == 0) qv[e] = -1;
    return;
  }
  int qn = (pidx < ES_N) ? sup_src[pidx] : sup_dst[pidx - ES_N];  // sup_query
  int sn = msg_src[e];
  float4 xa = *(const float4*)(nodes + (size_t)qn * CC + lane * 4);
  float4 xb = *(const float4*)(nodes + (size_t)sn * CC + lane * 4);
  float p = xa.x * xb.x + xa.y * xb.y + xa.z * xb.z + xa.w * xb.w;
  p = wave_sum(p);
  if (lane == 0) {
    float cs = p * inv[qn] * inv[sn];
    qv[e] = qn;
    cosv[e] = cs;
    atomicAdd(&denom[qn], __expf(10.0f * cs - 10.0f));
    atomicAdd(&cntf[qn], 1.0f);
  }
}

// numer[q] += softmax_weight * cos
__global__ __launch_bounds__(256) void k_wacc(const int* __restrict__ qv,
                                              const float* __restrict__ cosv,
                                              const float* __restrict__ denom,
                                              float* __restrict__ numer) {
  int e = blockIdx.x * 256 + threadIdx.x;
  if (e >= EM_N) return;
  int qn = qv[e];
  if (qn < 0) return;
  float c = cosv[e];
  float w = __expf(10.0f * c - 10.0f) / denom[qn];
  atomicAdd(&numer[qn], w * c);
}

__global__ __launch_bounds__(256) void k_cong(const float* __restrict__ numer,
                                              const float* __restrict__ cntf,
                                              float* __restrict__ cong) {
  int n = blockIdx.x * 256 + threadIdx.x;
  if (n >= NN) return;
  cong[n] = numer[n] / fmaxf(cntf[n], 1.0f);
}

// ---------------------------------------------------------------------------
// Fused main kernel: 16 support edges per block, 128 threads (4 waves).
//   combined=[a|m] (bf16 LDS) -> WMMA GEMM1 (K=256,N=256) -> LN+ReLU ->
//   WMMA GEMM2 (K=256,N=128) -> LN+ReLU -> heads (P_dec, P_nbr, P_cong, strengths)
// ---------------------------------------------------------------------------
__global__ __launch_bounds__(128) void k_main(
    const float* __restrict__ nodes, const float* __restrict__ nbrs,
    const int* __restrict__ sup_src, const int* __restrict__ sup_dst,
    const unsigned short* __restrict__ pkW0, const unsigned short* __restrict__ pkW1,
    const float* __restrict__ b0, const float* __restrict__ g0, const float* __restrict__ be0,
    const float* __restrict__ b1, const float* __restrict__ g1, const float* __restrict__ be1,
    const float* __restrict__ Wp, const float* __restrict__ bp,
    const float* __restrict__ Ww, const float* __restrict__ bw,
    const float* __restrict__ sim_scale, const float* __restrict__ sim_shift,
    const float* __restrict__ cong, const float* __restrict__ coeff,
    float* __restrict__ out) {
  __shared__ unsigned short xs[16][264];   // combined (bf16), K=256, padded stride
  __shared__ unsigned short hbs[16][264];  // hidden (bf16) after LN+ReLU
  __shared__ float hs[16][260];            // GEMM1 result (f32, pre-LN)
  __shared__ float efs[16][132];           // GEMM2 / edge features (f32)
  __shared__ float red[16][8][2];
  __shared__ float mu_s[16], rs_s[16];
  __shared__ int srcs[16], dsts[16];

  const int t = threadIdx.x;
  const int e0 = blockIdx.x * 16;
  const int w = t >> 5;
  const int l = t & 31;

  if (t < 16) { srcs[t] = sup_src[e0 + t]; dsts[t] = sup_dst[e0 + t]; }
  __syncthreads();

  {  // stage combined = [a | m] as bf16 (8 threads per row, 16 dims each)
    const int r = t >> 3, cp = t & 7;
    const float4* ps = (const float4*)(nodes + (size_t)srcs[r] * CC + cp * 16);
    const float4* pd = (const float4*)(nodes + (size_t)dsts[r] * CC + cp * 16);
#pragma unroll
    for (int v = 0; v < 4; ++v) {
      float4 x = ps[v], y = pd[v];
      int k = cp * 16 + v * 4;
      xs[r][k + 0] = f2bf(x.x + y.x); xs[r][CC + k + 0] = f2bf(x.x * y.x);
      xs[r][k + 1] = f2bf(x.y + y.y); xs[r][CC + k + 1] = f2bf(x.y * y.y);
      xs[r][k + 2] = f2bf(x.z + y.z); xs[r][CC + k + 2] = f2bf(x.z * y.z);
      xs[r][k + 3] = f2bf(x.w + y.w); xs[r][CC + k + 3] = f2bf(x.w * y.w);
    }
  }
  __syncthreads();

  // WMMA lane mapping (ISA 7.12.2)
  const int arow = l & 15;             // A: M row
  const int koff = (l < 16) ? 0 : 8;   // A: lane-half K offset
  const int colb = l & 15;             // C/D: N column
  const int mrow = (l >> 4) * 8;       // C/D: M base (0 or 8)

  // ---- GEMM1: [16 x 256] x [256 x 256], 16 n-tiles (4 per wave), 8 k-steps ----
  {
    v8f acc[4];
#pragma unroll
    for (int j = 0; j < 4; ++j) acc[j] = zero8();
#pragma unroll
    for (int ks = 0; ks < 8; ++ks) {
      const int k0 = ks * 32;
      v8u lo = *(const v8u*)&xs[arow][k0 + koff];
      v8u hi = *(const v8u*)&xs[arow][k0 + 16 + koff];
      BFV a;
      a.u = __builtin_shufflevector(lo, hi, 0, 1, 2, 3, 4, 5, 6, 7,
                                    8, 9, 10, 11, 12, 13, 14, 15);
#pragma unroll
      for (int j = 0; j < 4; ++j) {
        const int nt = w * 4 + j;
        BFV bm;
        bm.u = *(const v16u*)(pkW0 + ((size_t)(ks * 16 + nt) * 32 + l) * 16);
        acc[j] = __builtin_amdgcn_wmma_f32_16x16x32_bf16(
            false, a.b, false, bm.b, (short)0, acc[j], false, false);
      }
    }
#pragma unroll
    for (int j = 0; j < 4; ++j) {
      const int col = (w * 4 + j) * 16 + colb;
      const float bb = b0[col];
#pragma unroll
      for (int i = 0; i < 8; ++i) hs[mrow + i][col] = acc[j][i] + bb;
    }
  }
  __syncthreads();

  // ---- LayerNorm(256) + ReLU -> bf16 ----
  {
    const int r = t >> 3, p = t & 7;
    float s = 0.f, sq = 0.f;
#pragma unroll 4
    for (int i = 0; i < 32; ++i) { float v = hs[r][p * 32 + i]; s += v; sq += v * v; }
    red[r][p][0] = s; red[r][p][1] = sq;
  }
  __syncthreads();
  if (t < 16) {
    float s = 0.f, sq = 0.f;
#pragma unroll
    for (int p = 0; p < 8; ++p) { s += red[t][p][0]; sq += red[t][p][1]; }
    float mu = s * (1.0f / 256.0f);
    float var = sq * (1.0f / 256.0f) - mu * mu;
    mu_s[t] = mu; rs_s[t] = rsqrtf(var + 1e-5f);
  }
  __syncthreads();
#pragma unroll 4
  for (int i = 0; i < 32; ++i) {
    int flat = t * 32 + i;
    int r = flat >> 8, c = flat & 255;
    float v = (hs[r][c] - mu_s[r]) * rs_s[r] * g0[c] + be0[c];
    hbs[r][c] = f2bf(fmaxf(v, 0.f));
  }
  __syncthreads();

  // ---- GEMM2: [16 x 256] x [256 x 128], 8 n-tiles (2 per wave), 8 k-steps ----
  {
    v8f acc[2];
    acc[0] = zero8(); acc[1] = zero8();
#pragma unroll
    for (int ks = 0; ks < 8; ++ks) {
      const int k0 = ks * 32;
      v8u lo = *(const v8u*)&hbs[arow][k0 + koff];
      v8u hi = *(const v8u*)&hbs[arow][k0 + 16 + koff];
      BFV a;
      a.u = __builtin_shufflevector(lo, hi, 0, 1, 2, 3, 4, 5, 6, 7,
                                    8, 9, 10, 11, 12, 13, 14, 15);
#pragma unroll
      for (int j = 0; j < 2; ++j) {
        const int nt = w * 2 + j;
        BFV bm;
        bm.u = *(const v16u*)(pkW1 + ((size_t)(ks * 8 + nt) * 32 + l) * 16);
        acc[j] = __builtin_amdgcn_wmma_f32_16x16x32_bf16(
            false, a.b, false, bm.b, (short)0, acc[j], false, false);
      }
    }
#pragma unroll
    for (int j = 0; j < 2; ++j) {
      const int col = (w * 2 + j) * 16 + colb;
      const float bb = b1[col];
#pragma unroll
      for (int i = 0; i < 8; ++i) efs[mrow + i][col] = acc[j][i] + bb;
    }
  }
  __syncthreads();

  // ---- LayerNorm(128) + ReLU (in place, f32) ----
  {
    const int r = t >> 3, p = t & 7;
    float s = 0.f, sq = 0.f;
#pragma unroll
    for (int i = 0; i < 16; ++i) { float v = efs[r][p * 16 + i]; s += v; sq += v * v; }
    red[r][p][0] = s; red[r][p][1] = sq;
  }
  __syncthreads();
  if (t < 16) {
    float s = 0.f, sq = 0.f;
#pragma unroll
    for (int p = 0; p < 8; ++p) { s += red[t][p][0]; sq += red[t][p][1]; }
    float mu = s * (1.0f / 128.0f);
    float var = sq * (1.0f / 128.0f) - mu * mu;
    mu_s[t] = mu; rs_s[t] = rsqrtf(var + 1e-5f);
  }
  __syncthreads();
#pragma unroll
  for (int i = 0; i < 16; ++i) {
    int flat = t * 16 + i;
    int r = flat >> 7, c = flat & 127;
    float v = (efs[r][c] - mu_s[r]) * rs_s[r] * g1[c] + be1[c];
    efs[r][c] = fmaxf(v, 0.f);
  }
  __syncthreads();

  // ---- heads: one wave handles 4 rows, lane-parallel 128-d dots ----
  {
    const float c0 = coeff[0], c1 = coeff[1], c2 = coeff[2];
    const float ss = sim_scale[0], sh = sim_shift[0];
#pragma unroll
    for (int rr = 0; rr < 4; ++rr) {
      const int r = w * 4 + rr;
      const int e = e0 + r;
      const int sn = srcs[r], dn = dsts[r];
      float4 xa = *(const float4*)(nbrs + (size_t)sn * CC + l * 4);
      float4 xb = *(const float4*)(nbrs + (size_t)dn * CC + l * 4);
      float nsim = xa.x * xb.x + xa.y * xb.y + xa.z * xb.z + xa.w * xb.w;
      float pdec = 0.f, sw = 0.f;
#pragma unroll
      for (int j = 0; j < 4; ++j) {
        const int c = l * 4 + j;
        const float ef = efs[r][c];
        pdec += ef * Wp[c];
        sw += ef * Ww[c];
      }
      nsim = wave_sum(nsim);
      pdec = wave_sum(pdec);
      sw = wave_sum(sw);
      if (l == 0) {
        pdec += bp[0];
        float st = fmaxf(sw + nsim * Ww[CC] + bw[0], 0.f);
        float exist = cong[sn] + cong[dn];
        out[(size_t)e * 3 + 0] = c0 * pdec;               // coeff0 * P_dec
        out[(size_t)e * 3 + 1] = c1 * ss * (nsim + sh);   // coeff1 * P_nbr
        out[(size_t)e * 3 + 2] = c2 * ss * (exist + sh);  // coeff2 * P_cong
        out[(size_t)ES_N * 3 + e] = st;                   // strengths
      }
    }
  }
}

extern "C" void kernel_launch(void* const* d_in, const int* in_sizes, int n_in,
                              void* d_out, int out_size, void* d_ws, size_t ws_size,
                              hipStream_t stream) {
  (void)in_sizes; (void)n_in; (void)out_size; (void)ws_size;

  const float* nodes = (const float*)d_in[0];
  const float* nbrs = (const float*)d_in[1];
  const int* sup_src = (const int*)d_in[2];
  const int* sup_dst = (const int*)d_in[3];
  const int* msg_src = (const int*)d_in[4];
  const int* msg_dst = (const int*)d_in[5];
  // d_in[6] message_edgestr and d_in[22..27] mm_* are dead: they feed only
  // `strength`, which the reference deletes before returning.
  const float* W0 = (const float*)d_in[7];
  const float* b0 = (const float*)d_in[8];
  const float* g0 = (const float*)d_in[9];
  const float* be0 = (const float*)d_in[10];
  const float* W1 = (const float*)d_in[11];
  const float* b1 = (const float*)d_in[12];
  const float* g1 = (const float*)d_in[13];
  const float* be1 = (const float*)d_in[14];
  const float* Wp = (const float*)d_in[15];
  const float* bp = (const float*)d_in[16];
  const float* Ww = (const float*)d_in[17];
  const float* bw = (const float*)d_in[18];
  const float* sim_scale = (const float*)d_in[19];
  const float* sim_shift = (const float*)d_in[20];
  const float* coeff_raw = (const float*)d_in[21];
  float* out = (float*)d_out;

  // workspace carve-out (~9.4 MB total)
  char* p = (char*)d_ws;
  auto take = [&](size_t bytes) {
    char* q = p;
    p += (bytes + 255) & ~(size_t)255;
    return q;
  };
  float* inv = (float*)take((size_t)NN * 4);
  int* partner = (int*)take((size_t)NN * 4);
  int* qv = (int*)take((size_t)EM_N * 4);
  float* cosv = (float*)take((size_t)EM_N * 4);
  float* denom = (float*)take((size_t)(NN + 1) * 4);
  float* cntf = (float*)take((size_t)(NN + 1) * 4);
  float* numer = (float*)take((size_t)(NN + 1) * 4);
  float* cong = (float*)take((size_t)NN * 4);
  unsigned short* pk0 = (unsigned short*)take((size_t)65536 * 2);
  unsigned short* pk1 = (unsigned short*)take((size_t)32768 * 2);
  float* coeff = (float*)take(16);

  k_pack<<<65536 / 256, 256, 0, stream>>>(W0, H0_N, 16, 65536, pk0);
  k_pack<<<32768 / 256, 256, 0, stream>>>(W1, H1_N, 8, 32768, pk1);
  k_coeff<<<1, 32, 0, stream>>>(coeff_raw, coeff, out);
  k_init<<<(NN + 1 + 255) / 256, 256, 0, stream>>>(partner, denom, cntf, numer);
  k_unitinv<<<(NN * 32) / 256, 256, 0, stream>>>(nodes, inv);
  k_scatter<<<(2 * ES_N + 255) / 256, 256, 0, stream>>>(sup_src, sup_dst, partner);
  k_cos<<<EM_N / 8, 256, 0, stream>>>(nodes, inv, msg_src, msg_dst, partner,
                                      sup_src, sup_dst, qv, cosv, denom, cntf);
  k_wacc<<<(EM_N + 255) / 256, 256, 0, stream>>>(qv, cosv, denom, numer);
  k_cong<<<(NN + 255) / 256, 256, 0, stream>>>(numer, cntf, cong);
  k_main<<<ES_N / 16, 128, 0, stream>>>(nodes, nbrs, sup_src, sup_dst, pk0, pk1,
                                        b0, g0, be0, b1, g1, be1, Wp, bp, Ww, bw,
                                        sim_scale, sim_shift, cong, coeff, out);
}
